// FusionModule_26044681683353
// MI455X (gfx1250) — matrix operand order repaired and checked by reading, not compiled, not run on hardware
//
#include <hip/hip_runtime.h>
#include <hip/hip_bf16.h>

// ---------------------------------------------------------------------------
// Problem constants (fixed by the reference)
// ---------------------------------------------------------------------------
#define BATCH   2
#define EMBED   256
#define HH      128
#define WW      128
#define NPOS    (HH * WW)          // 16384
#define IN_DIM  512
#define NHEADS  8
#define NPTS    4
#define DHEAD   32
#define NPATCH  64                 // 8x8 patches of 16x16

typedef unsigned int  u32;
typedef unsigned short u16;
typedef __attribute__((ext_vector_type(4)))  u32    u32x4;
typedef __attribute__((ext_vector_type(16))) __bf16 bf16x16;
typedef __attribute__((ext_vector_type(8)))  float  f32x8;

__device__ __forceinline__ u16 f2bf(float f) {
    u32 u = __float_as_uint(f);
    u32 r = u + 0x7FFFu + ((u >> 16) & 1u);   // round-to-nearest-even
    return (u16)(r >> 16);
}

// ---------------------------------------------------------------------------
// K1: sinusoidal positional encoding, pe[c][n]  (c=0..255, n = h*128+w)
// ---------------------------------------------------------------------------
__global__ __launch_bounds__(256) void compute_pe(float* __restrict__ pe) {
    int t = blockIdx.x * 256 + threadIdx.x;          // 256*16384 threads
    if (t >= EMBED * NPOS) return;
    int n = t & (NPOS - 1);
    int c = t >> 14;
    int h = n >> 7, w = n & 127;
    int g = c >> 2, r = c & 3;
    // div = exp(2g * (-ln(10000)/128))
    float dv = __expf((float)(2 * g) * (-9.210340371976184f / 128.0f));
    float xx = (float)(w + 1) * dv;
    float yy = (float)(h + 1) * dv;
    float v = (r == 0) ? __sinf(xx) : (r == 1) ? __cosf(xx)
            : (r == 2) ? __sinf(yy) : __cosf(yy);
    pe[t] = v;
}

// ---------------------------------------------------------------------------
// K2: bilinear resize ground (B,512,32,32) -> bf16 [B, N, 512]
// half-pixel centers, edge clamp (matches jax.image.resize upsampling)
// ---------------------------------------------------------------------------
__global__ __launch_bounds__(256) void resize_pack(const float* __restrict__ g,
                                                   u16* __restrict__ dst) {
    size_t t = (size_t)blockIdx.x * 256 + threadIdx.x;   // B*N*512
    if (t >= (size_t)BATCH * NPOS * IN_DIM) return;
    int c = (int)(t & (IN_DIM - 1));
    int n = (int)((t >> 9) & (NPOS - 1));
    int b = (int)(t >> 23);
    int h = n >> 7, w = n & 127;
    float sy = ((float)h + 0.5f) * 0.25f - 0.5f;
    float sx = ((float)w + 0.5f) * 0.25f - 0.5f;
    float fy = floorf(sy), fx = floorf(sx);
    float ly = sy - fy,    lx = sx - fx;
    int y0 = (int)fy, x0 = (int)fx;
    int y0c = min(31, max(0, y0)),     x0c = min(31, max(0, x0));
    int y1c = min(31, max(0, y0 + 1)), x1c = min(31, max(0, x0 + 1));
    const float* gp = g + ((size_t)b * IN_DIM + c) * 1024;   // 32x32 plane
    float v = (1.f - ly) * ((1.f - lx) * gp[y0c * 32 + x0c] + lx * gp[y0c * 32 + x1c])
            +        ly  * ((1.f - lx) * gp[y1c * 32 + x0c] + lx * gp[y1c * 32 + x1c]);
    dst[t] = f2bf(v);
}

// ---------------------------------------------------------------------------
// K3: LDS-tiled transpose+pack: src f32 [B, C, N] -> dst bf16 [B, N, C]
// grid: (N/32, C/32, B), block 256
// ---------------------------------------------------------------------------
__global__ __launch_bounds__(256) void transpose_pack(const float* __restrict__ src,
                                                      u16* __restrict__ dst,
                                                      int C, int N) {
    __shared__ float tile[32][33];
    int n0 = blockIdx.x * 32, c0 = blockIdx.y * 32, b = blockIdx.z;
    int tx = threadIdx.x & 31;
    int ty = threadIdx.x >> 5;                  // 8 rows per pass
    const float* s = src + (size_t)b * C * N;
    #pragma unroll
    for (int i = 0; i < 32; i += 8)
        tile[ty + i][tx] = s[(size_t)(c0 + ty + i) * N + n0 + tx];
    __syncthreads();
    u16* d = dst + (size_t)b * N * C;
    #pragma unroll
    for (int i = 0; i < 32; i += 8)
        d[(size_t)(n0 + ty + i) * C + c0 + tx] = f2bf(tile[tx][ty + i]);
}

// ---------------------------------------------------------------------------
// K4: pack a weight matrix (f32, element (k,n) at W[k*sk + n*sn]) into the
// WMMA bf16 B-fragment layout: Bp[(kt*ntiles + nt)*256 + lane*8 + v] holds
// {B[k0],B[k0+1]} for k0 = kt*32 + (lane>>4)*16 + 2v, n = nt*16 + (lane&15).
// ---------------------------------------------------------------------------
__global__ __launch_bounds__(256) void pack_w(const float* __restrict__ W,
                                              u32* __restrict__ Bp,
                                              int K, int Nc, int sk, int sn) {
    int t = blockIdx.x * 256 + threadIdx.x;
    int total = (K * Nc) >> 1;
    if (t >= total) return;
    int v      = t & 7;
    int lane   = (t >> 3) & 31;
    int ntiles = Nc >> 4;
    int nt     = (t >> 8) % ntiles;
    int kt     = t / (ntiles << 8);
    int n  = nt * 16 + (lane & 15);
    int k0 = kt * 32 + (lane >> 4) * 16 + 2 * v;
    u32 lo = f2bf(W[(size_t)k0 * sk + (size_t)n * sn]);
    u32 hi = f2bf(W[(size_t)(k0 + 1) * sk + (size_t)n * sn]);
    Bp[t] = lo | (hi << 16);
}

// ---------------------------------------------------------------------------
// K5: tiled bf16 WMMA GEMM.  C[M,Nc] = A[M,K](bf16,row-major) x Bpacked + bias
// One wave computes a (16*MT) x (16*TPW) super-tile: MT A-fragments are each
// reused across TPW n-tiles, and every B fragment is reused across the MT
// m-tiles (register-level reuse: MT*TPW wmma per 2*(MT+TPW) b128 loads).
// mode 0: +bias.  mode 1: +bias + pe[n_channel][m&16383] (q build, dual out).
// mode 2: +bias + extra[m*Nc+n] (residual add).
// ---------------------------------------------------------------------------
template <int MT, int TPW>
__global__ __launch_bounds__(256) void gemm_bf16_wmma(
    const u16* __restrict__ A, const u32* __restrict__ Bp,
    const float* __restrict__ bias, const float* __restrict__ extra,
    float* __restrict__ Cf, u16* __restrict__ Cbf,
    int M, int K, int Nc, int mode) {
    int gw   = (blockIdx.x * 256 + threadIdx.x) >> 5;
    int lane = threadIdx.x & 31;
    int ntiles  = Nc >> 4;
    int nstrips = ntiles / TPW;
    int mt = gw / nstrips;
    int ns = gw % nstrips;
    if (mt * (16 * MT) >= M) return;             // wave-uniform: EXEC stays full
    int half = lane >> 4, l15 = lane & 15;
    const u16* Arow[MT];
    #pragma unroll
    for (int j = 0; j < MT; ++j)
        Arow[j] = A + (size_t)(mt * 16 * MT + j * 16 + l15) * K;
    f32x8 acc[MT][TPW] = {};
    int ktiles = K >> 5;
    for (int kt = 0; kt < ktiles; ++kt) {
        // A fragments: 16x32 bf16, per-lane two contiguous 16B chunks
        union { u32x4 u[2]; bf16x16 v; } ua[MT];
        #pragma unroll
        for (int j = 0; j < MT; ++j) {
            const u32x4* ap = (const u32x4*)(Arow[j] + kt * 32);
            ua[j].u[0] = ap[2 * half];
            ua[j].u[1] = ap[2 * half + 1];
            __builtin_prefetch(Arow[j] + (kt + 2) * 32, 0, 1);  // global_prefetch
        }
        const u32* bk = Bp + (size_t)kt * ntiles * 256;
        #pragma unroll
        for (int tt = 0; tt < TPW; ++tt) {
            union { u32x4 u[2]; bf16x16 v; } ub;
            const u32x4* bp4 = (const u32x4*)(bk + ((size_t)(ns * TPW + tt) * 32 + lane) * 8);
            ub.u[0] = bp4[0];
            ub.u[1] = bp4[1];
            #pragma unroll
            for (int j = 0; j < MT; ++j)
                acc[j][tt] = __builtin_amdgcn_wmma_f32_16x16x32_bf16(
                    false, ua[j].v, false, ub.v, (short)0, acc[j][tt], false, false);
        }
    }
    #pragma unroll
    for (int j = 0; j < MT; ++j) {
        #pragma unroll
        for (int tt = 0; tt < TPW; ++tt) {
            int n  = (ns * TPW + tt) * 16 + l15;
            float bv = bias ? bias[n] : 0.0f;
            #pragma unroll
            for (int r = 0; r < 8; ++r) {
                int m = mt * 16 * MT + j * 16 + r + half * 8;
                float v = acc[j][tt][r] + bv;
                if (mode == 1)      v += extra[(size_t)n * NPOS + (m & (NPOS - 1))];
                else if (mode == 2) v += extra[(size_t)m * Nc + n];
                size_t o = (size_t)m * Nc + n;
                if (Cf)  Cf[o]  = v;
                if (Cbf) Cbf[o] = f2bf(v);
            }
        }
    }
}

// ---------------------------------------------------------------------------
// K6: deformable attention sampling.
// block = one (b,n) query position, wave = head, lane = channel-in-head.
// val [B,N,256] f32 (channel = head*32+d), off [B*N,64], awlog [B*N,32].
// out samp bf16 [B,N,256].
// ---------------------------------------------------------------------------
__global__ __launch_bounds__(256) void msda_sample(const float* __restrict__ val,
                                                   const float* __restrict__ off,
                                                   const float* __restrict__ awlog,
                                                   u16* __restrict__ samp) {
    int bn = blockIdx.x;                 // 0 .. B*N-1
    int b  = bn >> 14;
    int n  = bn & (NPOS - 1);
    int hy = n >> 7, wx = n & 127;
    int head = threadIdx.x >> 5;
    int lane = threadIdx.x & 31;
    const float* offp = off   + (size_t)bn * (NHEADS * NPTS * 2) + head * 8;
    const float* awp  = awlog + (size_t)bn * (NHEADS * NPTS)     + head * 4;
    // softmax over the 4 points
    float l0 = awp[0], l1 = awp[1], l2 = awp[2], l3 = awp[3];
    float mx = fmaxf(fmaxf(l0, l1), fmaxf(l2, l3));
    float e0 = __expf(l0 - mx), e1 = __expf(l1 - mx);
    float e2 = __expf(l2 - mx), e3 = __expf(l3 - mx);
    float inv = 1.0f / (e0 + e1 + e2 + e3);
    float wp[4] = { e0 * inv, e1 * inv, e2 * inv, e3 * inv };
    float refx = (float)wx * (1.0f / 127.0f);
    float refy = (float)hy * (1.0f / 127.0f);
    const float* vb = val + (size_t)b * NPOS * EMBED + head * DHEAD + lane;
    float acc = 0.0f;
    #pragma unroll
    for (int p = 0; p < NPTS; ++p) {
        float x = (refx + offp[2 * p]     * (1.0f / 128.0f)) * 128.0f - 0.5f;
        float y = (refy + offp[2 * p + 1] * (1.0f / 128.0f)) * 128.0f - 0.5f;
        float xf = floorf(x), yf = floorf(y);
        int   x0 = (int)xf,   y0 = (int)yf;
        float lx = x - xf,    ly = y - yf;
        float s = 0.0f;
        #pragma unroll
        for (int dy = 0; dy < 2; ++dy) {
            #pragma unroll
            for (int dx = 0; dx < 2; ++dx) {
                int xi = x0 + dx, yi = y0 + dy;
                if (xi >= 0 && xi < WW && yi >= 0 && yi < HH) {      // zero pad
                    float wgt = (dx ? lx : 1.0f - lx) * (dy ? ly : 1.0f - ly);
                    s += wgt * vb[(size_t)(yi * WW + xi) * EMBED];
                }
            }
        }
        acc += wp[p] * s;
    }
    samp[(size_t)bn * EMBED + head * DHEAD + lane] = f2bf(acc);
}

// ---------------------------------------------------------------------------
// K7: wave-per-row LayerNorm over 256 channels, in place.  x [B*N, 256]
// ---------------------------------------------------------------------------
__global__ __launch_bounds__(256) void layernorm_inplace(float* __restrict__ x,
                                                         const float* __restrict__ gamma,
                                                         const float* __restrict__ beta) {
    int row  = (blockIdx.x * 256 + threadIdx.x) >> 5;
    int lane = threadIdx.x & 31;
    if (row >= BATCH * NPOS) return;
    float* p = x + (size_t)row * EMBED + lane * 8;
    float v[8], s = 0.0f, s2 = 0.0f;
    #pragma unroll
    for (int i = 0; i < 8; ++i) { v[i] = p[i]; s += v[i]; }
    #pragma unroll
    for (int o = 16; o >= 1; o >>= 1) s += __shfl_xor(s, o, 32);
    float mu = s * (1.0f / 256.0f);
    #pragma unroll
    for (int i = 0; i < 8; ++i) { float d = v[i] - mu; s2 += d * d; }
    #pragma unroll
    for (int o = 16; o >= 1; o >>= 1) s2 += __shfl_xor(s2, o, 32);
    float rinv = rsqrtf(s2 * (1.0f / 256.0f) + 1e-5f);
    #pragma unroll
    for (int i = 0; i < 8; ++i) {
        int c = lane * 8 + i;
        p[i] = (v[i] - mu) * rinv * gamma[c] + beta[c];
    }
}

// ---------------------------------------------------------------------------
// K8: per-patch routing logits + softmax.  One block per (b,patch).
// feature index e = c*256 + ph*16 + pw ; logits scaled by 1/sqrt(65536)=1/256.
// ---------------------------------------------------------------------------
__global__ __launch_bounds__(256) void route_logits(const float* __restrict__ p1,
                                                    const float* __restrict__ p2,
                                                    const float* __restrict__ Wl,
                                                    const float* __restrict__ bl,
                                                    float* __restrict__ wts,
                                                    float* __restrict__ out_w) {
    int b = blockIdx.x >> 6;
    int patch = blockIdx.x & 63;
    int pr = patch >> 3, pc = patch & 7;
    int tid = threadIdx.x;                    // tid = channel c
    float a0 = 0.0f, a1 = 0.0f;
    for (int nl = 0; nl < 256; ++nl) {        // position within patch
        int ph = nl >> 4, pw = nl & 15;
        int n  = (pr * 16 + ph) * WW + pc * 16 + pw;
        size_t idx = ((size_t)b * NPOS + n) * EMBED + tid;
        float x = 0.5f * (p1[idx] + p2[idx]);
        int e = tid * 256 + nl;
        a0 += x * Wl[(size_t)e * 2];
        a1 += x * Wl[(size_t)e * 2 + 1];
    }
    __shared__ float r0[256], r1[256];
    r0[tid] = a0; r1[tid] = a1;
    __syncthreads();
    for (int s = 128; s > 0; s >>= 1) {
        if (tid < s) { r0[tid] += r0[tid + s]; r1[tid] += r1[tid + s]; }
        __syncthreads();
    }
    if (tid == 0) {
        float q0 = (r0[0] + bl[0]) * (1.0f / 256.0f);
        float q1 = (r1[0] + bl[1]) * (1.0f / 256.0f);
        float m = fmaxf(q0, q1);
        float f0 = __expf(q0 - m), f1 = __expf(q1 - m);
        float iv = 1.0f / (f0 + f1);
        int o = (b * NPATCH + patch) * 2;
        wts[o] = f0 * iv;  wts[o + 1] = f1 * iv;
        out_w[o] = f0 * iv; out_w[o + 1] = f1 * iv;
    }
}

// ---------------------------------------------------------------------------
// K9: routed combine + transpose back to [B, C, H, W] via LDS tiles.
// grid (N/32, C/32, B), block 256.
// ---------------------------------------------------------------------------
__global__ __launch_bounds__(256) void route_combine(const float* __restrict__ p1,
                                                     const float* __restrict__ p2,
                                                     const float* __restrict__ wts,
                                                     float* __restrict__ out) {
    __shared__ float tile[32][33];
    int n0 = blockIdx.x * 32, c0 = blockIdx.y * 32, b = blockIdx.z;
    int tx = threadIdx.x & 31;
    int ty = threadIdx.x >> 5;
    const float* q1 = p1 + (size_t)b * NPOS * EMBED;
    const float* q2 = p2 + (size_t)b * NPOS * EMBED;
    #pragma unroll
    for (int i = 0; i < 32; i += 8) {
        int n = n0 + ty + i;
        int h = n >> 7, w = n & 127;
        int patch = (h >> 4) * 8 + (w >> 4);
        float w0 = wts[(b * NPATCH + patch) * 2];
        float w1 = wts[(b * NPATCH + patch) * 2 + 1];
        size_t idx = (size_t)n * EMBED + c0 + tx;
        tile[ty + i][tx] = w0 * q1[idx] + w1 * q2[idx];
    }
    __syncthreads();
    float* o = out + (size_t)b * EMBED * NPOS;
    #pragma unroll
    for (int i = 0; i < 32; i += 8) {
        int c = c0 + ty + i;
        o[(size_t)c * NPOS + n0 + tx] = tile[tx][ty + i];
    }
}

// ---------------------------------------------------------------------------
// Host-side orchestration
// ---------------------------------------------------------------------------
static inline void launch_gemm(const u16* A, const u32* Bp, const float* bias,
                               const float* extra, float* Cf, u16* Cbf,
                               int M, int K, int Nc, int mode, hipStream_t s) {
    int ntiles = Nc >> 4;
    // All GEMMs here have M % 32 == 0, so MT=2 always applies.
    if ((ntiles & 3) == 0) {
        long waves = (long)(M / 32) * (ntiles / 4);
        int blocks = (int)((waves * 32 + 255) / 256);
        gemm_bf16_wmma<2, 4><<<blocks, 256, 0, s>>>(A, Bp, bias, extra, Cf, Cbf, M, K, Nc, mode);
    } else {
        long waves = (long)(M / 32) * (ntiles / 2);
        int blocks = (int)((waves * 32 + 255) / 256);
        gemm_bf16_wmma<2, 2><<<blocks, 256, 0, s>>>(A, Bp, bias, extra, Cf, Cbf, M, K, Nc, mode);
    }
}

static inline void launch_pack_w(const float* W, u32* Bp, int K, int Nc,
                                 int sk, int sn, hipStream_t s) {
    int total = (K * Nc) >> 1;
    pack_w<<<(total + 255) / 256, 256, 0, s>>>(W, Bp, K, Nc, sk, sn);
}

extern "C" void kernel_launch(void* const* d_in, const int* in_sizes, int n_in,
                              void* d_out, int out_size, void* d_ws, size_t ws_size,
                              hipStream_t stream) {
    const float* ground = (const float*)d_in[0];
    const float* imgs[2] = { (const float*)d_in[1], (const float*)d_in[2] };   // sat, osm
    const float* conv_w = (const float*)d_in[3];
    const float* conv_b = (const float*)d_in[4];
    // per-branch: Wv bv Wo bo Wa ba Wp bp  at indices 5..12 (sat), 13..20 (osm)
    const float* Wb[2][8];
    for (int br = 0; br < 2; ++br)
        for (int i = 0; i < 8; ++i) Wb[br][i] = (const float*)d_in[5 + br * 8 + i];
    const float* ln_g[2] = { (const float*)d_in[21], (const float*)d_in[23] };
    const float* ln_b[2] = { (const float*)d_in[22], (const float*)d_in[24] };
    const float* Wl = (const float*)d_in[25];
    const float* bl = (const float*)d_in[26];

    float* out_img = (float*)d_out;                          // [B,256,128,128]
    float* out_wts = out_img + (size_t)BATCH * EMBED * NPOS; // [B,64,2]

    // ---- workspace layout (~198 MB, with region reuse) ----
    char* ws = (char*)d_ws;
    size_t off = 0;
    auto alloc = [&](size_t bytes) -> char* {
        char* p = ws + off;
        off += (bytes + 255) & ~(size_t)255;
        return p;
    };
    float* pe     = (float*)alloc((size_t)EMBED * NPOS * 4);                 // 16.78 MB
    u16*   gr_bf  = (u16*)  alloc((size_t)BATCH * NPOS * IN_DIM * 2);        // 33.55 MB
    float* q_f32  = (float*)alloc((size_t)BATCH * NPOS * EMBED * 4);         // 33.55 MB
    u16*   q_bf   = (u16*)  alloc((size_t)BATCH * NPOS * EMBED * 2);         // 16.78 MB
    u16*   v_bf   = (u16*)  alloc((size_t)BATCH * NPOS * EMBED * 2);         // 16.78 MB
    float* off_b  = (float*)alloc((size_t)BATCH * NPOS * 64 * 4);            //  8.39 MB
    float* aw_b   = (float*)alloc((size_t)BATCH * NPOS * 32 * 4);            //  4.19 MB
    float* outb0  = (float*)alloc((size_t)BATCH * NPOS * EMBED * 4);         // 33.55 MB
    float* outb1  = (float*)alloc((size_t)BATCH * NPOS * EMBED * 4);         // 33.55 MB
    u32* convWp   = (u32*)alloc((size_t)IN_DIM * EMBED * 2);
    u32* Wv_p[2], *Wo_p[2], *Wa_p[2], *Wp_p[2];
    for (int br = 0; br < 2; ++br) {
        Wv_p[br] = (u32*)alloc((size_t)EMBED * EMBED * 2);
        Wo_p[br] = (u32*)alloc((size_t)EMBED * 64 * 2);
        Wa_p[br] = (u32*)alloc((size_t)EMBED * 32 * 2);
        Wp_p[br] = (u32*)alloc((size_t)EMBED * EMBED * 2);
    }
    float* wts = (float*)alloc((size_t)BATCH * NPATCH * 2 * 4);
    if (off > ws_size) return;   // workspace too small: bail deterministically

    // Region reuse (safe by kernel ordering):
    //   gr_bf (consumed by q-GEMM)   -> val  (per-branch value tensor, f32)
    //   pe    (consumed by q-GEMM)   -> samp (per-branch sampled bf16)
    float* val  = (float*)gr_bf;
    u16*   samp = (u16*)pe;

    // 1) positional encoding
    compute_pe<<<(EMBED * NPOS) / 256, 256, 0, stream>>>(pe);
    // 2) upsample + bf16 pack of ground
    {
        size_t tot = (size_t)BATCH * NPOS * IN_DIM;
        resize_pack<<<(int)((tot + 255) / 256), 256, 0, stream>>>(ground, gr_bf);
    }
    // 3) pack all weight matrices into WMMA B layout
    //    conv_w is (o,c) used as g[c]*W[o,c]  -> B(k=c,n=o): sk=1, sn=512
    launch_pack_w(conv_w, convWp, IN_DIM, EMBED, 1, IN_DIM, stream);
    for (int br = 0; br < 2; ++br) {
        launch_pack_w(Wb[br][0], Wv_p[br], EMBED, EMBED, EMBED, 1, stream);  // Wv
        launch_pack_w(Wb[br][2], Wo_p[br], EMBED, 64,    64,    1, stream);  // Wo
        launch_pack_w(Wb[br][4], Wa_p[br], EMBED, 32,    32,    1, stream);  // Wa
        launch_pack_w(Wb[br][6], Wp_p[br], EMBED, EMBED, EMBED, 1, stream);  // Wp
    }
    // 4) q = resize(ground) x conv_w + conv_b + pe   (mode 1, dual f32/bf16)
    launch_gemm(gr_bf, convWp, conv_b, pe, q_f32, q_bf,
                BATCH * NPOS, IN_DIM, EMBED, 1, stream);

    // 5) per-branch deformable cross attention
    float* outb[2] = { outb0, outb1 };
    for (int br = 0; br < 2; ++br) {
        // value image -> bf16 [B,N,256]
        transpose_pack<<<dim3(NPOS / 32, EMBED / 32, BATCH), 256, 0, stream>>>(
            imgs[br], v_bf, EMBED, NPOS);
        // val = v x Wv + bv
        launch_gemm(v_bf, Wv_p[br], Wb[br][1], nullptr, val, nullptr,
                    BATCH * NPOS, EMBED, EMBED, 0, stream);
        // off = q x Wo + bo   (64 cols)
        launch_gemm(q_bf, Wo_p[br], Wb[br][3], nullptr, off_b, nullptr,
                    BATCH * NPOS, EMBED, 64, 0, stream);
        // aw logits = q x Wa + ba (32 cols)
        launch_gemm(q_bf, Wa_p[br], Wb[br][5], nullptr, aw_b, nullptr,
                    BATCH * NPOS, EMBED, 32, 0, stream);
        // bilinear deformable sampling (+ per-head softmax over points)
        msda_sample<<<BATCH * NPOS, 256, 0, stream>>>(val, off_b, aw_b, samp);
        // proj: out = samp x Wp + bp + q   (mode 2 residual)
        launch_gemm(samp, Wp_p[br], Wb[br][7], q_f32, outb[br], nullptr,
                    BATCH * NPOS, EMBED, EMBED, 2, stream);
        // LayerNorm over channels, in place
        layernorm_inplace<<<(BATCH * NPOS * 32) / 256, 256, 0, stream>>>(
            outb[br], ln_g[br], ln_b[br]);
    }

    // 6) patch routing weights
    route_logits<<<BATCH * NPATCH, 256, 0, stream>>>(outb0, outb1, Wl, bl, wts, out_wts);
    // 7) routed combine -> [B, C, H, W]
    route_combine<<<dim3(NPOS / 32, EMBED / 32, BATCH), 256, 0, stream>>>(
        outb0, outb1, wts, out_img);
}
